// STP_G_Net_1202590843137
// MI455X (gfx1250) — matrix operand order repaired and checked by reading, not compiled, not run on hardware
//
#include <hip/hip_runtime.h>
#include <hip/hip_bf16.h>
#include <float.h>

// ---------------------------------------------------------------------------
// Types / constants
// ---------------------------------------------------------------------------
typedef __attribute__((ext_vector_type(16))) __bf16 v16bf;
typedef __attribute__((ext_vector_type(8)))  float  v8f;

#define H_ENC   128
#define HEADS   3
#define H_DEC   256
#define OUT_LEN 25
#define HIST    16
#define D_EMB   32

// ---------------------------------------------------------------------------
// Small device helpers
// ---------------------------------------------------------------------------
__device__ inline unsigned short f2us(float f) {
  unsigned u = __float_as_uint(f);
  unsigned r = (u + 0x7fffu + ((u >> 16) & 1u)) >> 16;   // round-nearest-even
  return (unsigned short)r;
}
__device__ inline __bf16 f2bf(float f) { return __builtin_bit_cast(__bf16, f2us(f)); }
__device__ inline float  bf2f(__bf16 b) {
  unsigned u = ((unsigned)__builtin_bit_cast(unsigned short, b)) << 16;
  return __uint_as_float(u);
}
__device__ inline float leakyf(float x, float s) { return x >= 0.f ? x : s * x; }
__device__ inline float sigmf(float x) { return 1.f / (1.f + __expf(-x)); }
// monotonic int encoding of float for atomicMax-based segment max
__device__ inline int   f2ord(float f) { int i = __float_as_int(f); return i >= 0 ? i : (i ^ 0x7fffffff); }
__device__ inline float ord2f(int i)   { return __int_as_float(i >= 0 ? i : (i ^ 0x7fffffff)); }

// A-fragment (16x32 bf16, M x K) loaded from an LDS row base at k0.
// lanes 0-15: K 0-7 & 16-23 ; lanes 16-31: K 8-15 & 24-31  (ah = 0 or 8)
__device__ inline v16bf afrag(const __bf16* rb, int ah) {
  v16bf r;
#pragma unroll
  for (int i = 0; i < 8; ++i) { r[i] = rb[ah + i]; r[8 + i] = rb[ah + 16 + i]; }
  return r;
}

// Packed-B layout: weights pre-packed into per-lane fragment order.
// Fragment (kb, nb) for lane L is 16 contiguous bf16 (32 B, aligned) at:
//   Bp + (((kb * nTiles + nb) * 32 + L) * 16)
__device__ inline const __bf16* bptr(const __bf16* __restrict__ Bp, int nTiles,
                                     int kb, int nb, int lane) {
  return Bp + ((((size_t)kb * nTiles + nb) * 32 + lane) << 4);
}
__device__ inline v16bf bfragp(const __bf16* __restrict__ p) {
  return *(const v16bf*)p;   // 32-byte aligned contiguous -> global_load_b128 x2
}

// ---------------------------------------------------------------------------
// Pack fp32 weight [K,N] row-major -> bf16 WMMA-fragment order (see bptr).
// ---------------------------------------------------------------------------
__global__ void pack_b_kernel(const float* __restrict__ src, __bf16* __restrict__ dst,
                              int K, int N) {
  int idx = blockIdx.x * 256 + threadIdx.x;
  if (idx >= K * N) return;
  int i    = idx & 15;
  int lane = (idx >> 4) & 31;
  int tile = idx >> 9;
  int nTiles = N >> 4;
  int nb = tile % nTiles;
  int kb = tile / nTiles;
  int k = kb * 32 + ((lane < 16) ? 0 : 8) + ((i < 8) ? i : 8 + i);
  int n = nb * 16 + (lane & 15);
  dst[idx] = f2bf(src[(size_t)k * N + n]);
}

// ---------------------------------------------------------------------------
// Generic bf16 WMMA GEMM: C[M,N] = act(A[M,K] @ B[K,N] + bias), B pre-packed.
// Block: 256 threads = 8 waves; each wave one 16-col tile; grid=(ceil(M/16), N/128)
// ---------------------------------------------------------------------------
template <int ACT>  // 0 = none, 1 = leaky(0.1)
__global__ __launch_bounds__(256) void gemm_bf16(
    const __bf16* __restrict__ A, int lda,
    const __bf16* __restrict__ Bp,
    const float* __restrict__ bias,
    float* __restrict__ Cf, __bf16* __restrict__ Cbf,
    int M, int N, int K) {
  __shared__ __bf16 At[16 * 384];           // K <= 384
  const int tid = threadIdx.x;
  const int mbase = blockIdx.x * 16;
  for (int idx = tid; idx < 16 * K; idx += 256) {
    int r = idx / K, c = idx - r * K;
    int gm = mbase + r;
    At[r * K + c] = (gm < M) ? A[(size_t)gm * lda + c] : f2bf(0.f);
  }
  __syncthreads();

  const int wave = tid >> 5, lane = tid & 31;
  const int nTiles = N >> 4;
  const int nb = blockIdx.y * 8 + wave;
  const int ah = (lane < 16) ? 0 : 8;
  const int ncol = nb * 16 + (lane & 15);
  const int ml = (lane < 16) ? 0 : 8;

  __builtin_prefetch(bptr(Bp, nTiles, 0, nb, lane), 0, 0);   // global_prefetch_b8

  v8f acc = {};
  for (int k0 = 0; k0 < K; k0 += 32) {
    v16bf af = afrag(&At[(lane & 15) * K + k0], ah);
    v16bf bf_ = bfragp(bptr(Bp, nTiles, k0 >> 5, nb, lane));
    acc = __builtin_amdgcn_wmma_f32_16x16x32_bf16(false, af, false, bf_, (short)0, acc, false, false);
  }
  float bv = bias ? bias[ncol] : 0.f;
#pragma unroll
  for (int v = 0; v < 8; ++v) {
    int gm = mbase + ml + v;
    if (gm < M) {
      float val = acc[v] + bv;
      if (ACT == 1) val = leakyf(val, 0.1f);
      if (Cf)  Cf[(size_t)gm * N + ncol] = val;
      if (Cbf) Cbf[(size_t)gm * N + ncol] = f2bf(val);
    }
  }
}

// ---------------------------------------------------------------------------
// Fused input-embedding + GRU encoder. One workgroup = 16 nodes, 16 steps.
// h state lives in LDS (fp32 + bf16 mirror); gate matmuls via WMMA bf16.
// Emits leaky(h_final, 0.1) as bf16 (input to dyn_emb GEMM).
// ---------------------------------------------------------------------------
__global__ __launch_bounds__(256) void gru_kernel(
    const float* __restrict__ x,       // [N,16,2]
    const float* __restrict__ ipW,     // [2,32]
    const float* __restrict__ ipb,     // [32]
    const __bf16* __restrict__ Wx,     // packed [32,384]
    const __bf16* __restrict__ Wh,     // packed [128,384]
    const float* __restrict__ bx,      // [384]
    const float* __restrict__ bh,      // [384]
    __bf16* __restrict__ hleak,        // [N,128]
    int N) {
  __shared__ __bf16 emb[16 * 32];
  __shared__ __bf16 hbf[16 * 128];
  __shared__ float  hf[16 * 128];
  __shared__ float  gx[16 * 384];
  __shared__ float  gh[16 * 384];

  const int tid = threadIdx.x;
  const int wave = tid >> 5, lane = tid & 31;
  const int mbase = blockIdx.x * 16;
  const int ah = (lane < 16) ? 0 : 8;
  const int nlo = lane & 15;
  const int ml = (lane < 16) ? 0 : 8;
  const int NT = 24;                     // 384/16 column tiles

  for (int i = tid; i < 16 * 128; i += 256) { hf[i] = 0.f; hbf[i] = f2bf(0.f); }
  __builtin_prefetch(bptr(Wh, NT, 0, wave * 3, lane), 0, 0);
  __syncthreads();

  for (int t = 0; t < HIST; ++t) {
    // leaky(ip_emb)
    for (int i = tid; i < 16 * 32; i += 256) {
      int m = i >> 5, d = i & 31;
      int gm = mbase + m;
      float v = 0.f;
      if (gm < N) {
        const float* xp = x + (size_t)gm * HIST * 2 + t * 2;
        v = leakyf(xp[0] * ipW[d] + xp[1] * ipW[32 + d] + ipb[d], 0.1f);
      }
      emb[i] = f2bf(v);
    }
    __syncthreads();
    // gate pre-activations: gx = emb@Wx + bx ; gh = h@Wh + bh
    for (int c = 0; c < 3; ++c) {
      int nb = wave * 3 + c;
      int ncol = nb * 16 + nlo;
      v8f ax = {};
      {
        v16bf af = afrag(&emb[(lane & 15) * 32], ah);
        v16bf bf_ = bfragp(bptr(Wx, NT, 0, nb, lane));
        ax = __builtin_amdgcn_wmma_f32_16x16x32_bf16(false, af, false, bf_, (short)0, ax, false, false);
      }
      v8f ahh = {};
      for (int k0 = 0; k0 < 128; k0 += 32) {
        v16bf af = afrag(&hbf[(lane & 15) * 128 + k0], ah);
        v16bf bf_ = bfragp(bptr(Wh, NT, k0 >> 5, nb, lane));
        ahh = __builtin_amdgcn_wmma_f32_16x16x32_bf16(false, af, false, bf_, (short)0, ahh, false, false);
      }
      float bxv = bx[ncol], bhv = bh[ncol];
#pragma unroll
      for (int v = 0; v < 8; ++v) {
        int mloc = ml + v;
        gx[mloc * 384 + ncol] = ax[v] + bxv;
        gh[mloc * 384 + ncol] = ahh[v] + bhv;
      }
    }
    __syncthreads();
    // GRU update
    for (int i = tid; i < 16 * 128; i += 256) {
      int m = i >> 7, j = i & 127;
      float r = sigmf(gx[m * 384 + j] + gh[m * 384 + j]);
      float z = sigmf(gx[m * 384 + 128 + j] + gh[m * 384 + 128 + j]);
      float nn = tanhf(gx[m * 384 + 256 + j] + r * gh[m * 384 + 256 + j]);
      float h = (1.f - z) * nn + z * hf[i];
      hf[i] = h; hbf[i] = f2bf(h);
    }
    __syncthreads();
  }
  for (int i = tid; i < 16 * 128; i += 256) {
    int m = i >> 7, j = i & 127;
    int gm = mbase + m;
    if (gm < N) hleak[(size_t)gm * 128 + j] = f2bf(leakyf(hf[i], 0.1f));
  }
}

// ---------------------------------------------------------------------------
// GAT attention kernels
// ---------------------------------------------------------------------------
__global__ void gat_scores_kernel(const float* __restrict__ hfeat,
                                  const float* __restrict__ as_, const float* __restrict__ ad_,
                                  float* __restrict__ ssrc, float* __restrict__ sdst, int N) {
  int gid = blockIdx.x * 256 + threadIdx.x;
  if (gid >= N * HEADS) return;
  int n = gid / HEADS, h = gid - n * HEADS;
  const float* hp = hfeat + (size_t)n * (HEADS * H_ENC) + h * H_ENC;
  float s0 = 0.f, s1 = 0.f;
  for (int c = 0; c < H_ENC; ++c) { float v = hp[c]; s0 += v * as_[h * H_ENC + c]; s1 += v * ad_[h * H_ENC + c]; }
  ssrc[gid] = s0; sdst[gid] = s1;
}

__global__ void fill_ord_kernel(int* __restrict__ p, float v, int n) {
  int i = blockIdx.x * 256 + threadIdx.x;
  if (i < n) p[i] = f2ord(v);
}

__device__ inline void edge_sd(const int* ei, int E, int e, int& s, int& d) {
  if (e < E) { s = ei[e]; d = ei[E + e]; } else { s = d = e - E; }  // self-loops appended
}

__global__ void edge_max_kernel(const int* __restrict__ ei, int E, int ET,
                                const float* __restrict__ ssrc, const float* __restrict__ sdst,
                                int* __restrict__ maxb) {
  int gid = blockIdx.x * 256 + threadIdx.x;
  if (gid >= ET * HEADS) return;
  int e = gid / HEADS, h = gid - e * HEADS;
  int s, d; edge_sd(ei, E, e, s, d);
  float v = leakyf(ssrc[s * HEADS + h] + sdst[d * HEADS + h], 0.2f);
  atomicMax(&maxb[d * HEADS + h], f2ord(v));
}

__global__ void edge_exp_kernel(const int* __restrict__ ei, int E, int ET,
                                const float* __restrict__ ssrc, const float* __restrict__ sdst,
                                const int* __restrict__ maxb,
                                float* __restrict__ exbuf, float* __restrict__ denom) {
  int gid = blockIdx.x * 256 + threadIdx.x;
  if (gid >= ET * HEADS) return;
  int e = gid / HEADS, h = gid - e * HEADS;
  int s, d; edge_sd(ei, E, e, s, d);
  float v = leakyf(ssrc[s * HEADS + h] + sdst[d * HEADS + h], 0.2f);
  float ex = __expf(v - ord2f(maxb[d * HEADS + h]));
  exbuf[gid] = ex;
  atomicAdd(&denom[d * HEADS + h], ex);
}

__global__ void edge_msg_kernel(const int* __restrict__ ei, int E, int ET,
                                const float* __restrict__ hfeat,
                                const float* __restrict__ exbuf, const float* __restrict__ denom,
                                float* __restrict__ outp) {
  int gid = blockIdx.x * 256 + threadIdx.x;
  if (gid >= ET * HEADS) return;
  int e = gid / HEADS, h = gid - e * HEADS;
  int s, d; edge_sd(ei, E, e, s, d);
  float alpha = exbuf[gid] / denom[d * HEADS + h];
  const float* hp = hfeat + (size_t)s * (HEADS * H_ENC) + h * H_ENC;
  float* op = outp + (size_t)d * (HEADS * H_ENC) + h * H_ENC;
  for (int c = 0; c < H_ENC; ++c) atomicAdd(&op[c], hp[c] * alpha);
}

__global__ void bias_bf_kernel(float* __restrict__ io, const float* __restrict__ b,
                               __bf16* __restrict__ obf, int total, int C) {
  int i = blockIdx.x * 256 + threadIdx.x;
  if (i >= total) return;
  float v = io[i] + b[i % C];
  io[i] = v;
  if (obf) obf[i] = f2bf(v);
}

__global__ void gather_pad_kernel(const float* __restrict__ src, const int* __restrict__ tgt,
                                  __bf16* __restrict__ dst, int T, int Mp, int C) {
  int i = blockIdx.x * 256 + threadIdx.x;
  if (i >= Mp * C) return;
  int m = i / C, c = i - m * C;
  float v = (m < T) ? src[(size_t)tgt[m] * C + c] : 0.f;
  dst[i] = f2bf(v);
}

// ---------------------------------------------------------------------------
// Fused 2-layer LSTM decoder. One workgroup = 16 targets, 25 steps.
// All state + 16x1024 gate buffer in dynamic LDS (112 KB < 320 KB/WGP).
// Output head fused (h1 @ op_W + op_b).
// ---------------------------------------------------------------------------
__global__ __launch_bounds__(256) void decoder_kernel(
    const float* __restrict__ gx0,     // [Mp,1024] = enc@l0_Wx + l0_b (constant/step)
    const __bf16* __restrict__ Wh0,    // packed [256,1024]
    const __bf16* __restrict__ Wx1,    // packed [256,1024]
    const __bf16* __restrict__ Wh1,    // packed [256,1024]
    const float* __restrict__ b1,      // [1024]
    const float* __restrict__ opW,     // [256,2]
    const float* __restrict__ opb,     // [2]
    float* __restrict__ out,           // [M,25,2]
    int M) {
  extern __shared__ char smem[];
  float*  g  = (float*)smem;                  // 16*1024 fp32 (64 KB)
  float*  c0 = g + 16 * 1024;                 // 16*256
  float*  c1 = c0 + 16 * 256;                 // 16*256
  __bf16* h0 = (__bf16*)(c1 + 16 * 256);      // 16*256
  __bf16* h1 = h0 + 16 * 256;                 // 16*256

  const int tid = threadIdx.x;
  const int wave = tid >> 5, lane = tid & 31;
  const int mbase = blockIdx.x * 16;
  const int ah = (lane < 16) ? 0 : 8;
  const int nlo = lane & 15;
  const int ml = (lane < 16) ? 0 : 8;
  const int NT = 64;                          // 1024/16 column tiles

  for (int i = tid; i < 16 * 256; i += 256) {
    c0[i] = 0.f; c1[i] = 0.f; h0[i] = f2bf(0.f); h1[i] = f2bf(0.f);
  }
  __builtin_prefetch(bptr(Wh0, NT, 0, wave * 8, lane), 0, 0);
  __builtin_prefetch(bptr(Wh1, NT, 0, wave * 8, lane), 0, 0);
  __syncthreads();

  for (int t = 0; t < OUT_LEN; ++t) {
    // ---- layer0 gates: g = h0 @ Wh0 + gx0 ----
    for (int c = 0; c < 8; ++c) {
      int nb = wave * 8 + c;
      int ncol = nb * 16 + nlo;
      v8f acc = {};
      for (int k0 = 0; k0 < 256; k0 += 32) {
        v16bf af = afrag(&h0[(lane & 15) * 256 + k0], ah);
        v16bf bf_ = bfragp(bptr(Wh0, NT, k0 >> 5, nb, lane));
        acc = __builtin_amdgcn_wmma_f32_16x16x32_bf16(false, af, false, bf_, (short)0, acc, false, false);
      }
#pragma unroll
      for (int v = 0; v < 8; ++v) {
        int mloc = ml + v;
        g[mloc * 1024 + ncol] = acc[v] + gx0[(size_t)(mbase + mloc) * 1024 + ncol];
      }
    }
    __syncthreads();
    // ---- h0/c0 update ----
    for (int i = tid; i < 16 * 256; i += 256) {
      int m = i >> 8, j = i & 255;
      float gi = sigmf(g[m * 1024 + j]);
      float gf = sigmf(g[m * 1024 + 256 + j]);
      float gg = tanhf(g[m * 1024 + 512 + j]);
      float go = sigmf(g[m * 1024 + 768 + j]);
      float c = gf * c0[i] + gi * gg;
      c0[i] = c;
      h0[i] = f2bf(go * tanhf(c));
    }
    __syncthreads();
    // ---- layer1 gates: g = h0 @ Wx1 + h1 @ Wh1 + b1 ----
    for (int c = 0; c < 8; ++c) {
      int nb = wave * 8 + c;
      int ncol = nb * 16 + nlo;
      v8f acc = {};
      for (int k0 = 0; k0 < 256; k0 += 32) {
        v16bf af = afrag(&h0[(lane & 15) * 256 + k0], ah);
        v16bf bf_ = bfragp(bptr(Wx1, NT, k0 >> 5, nb, lane));
        acc = __builtin_amdgcn_wmma_f32_16x16x32_bf16(false, af, false, bf_, (short)0, acc, false, false);
      }
      for (int k0 = 0; k0 < 256; k0 += 32) {
        v16bf af = afrag(&h1[(lane & 15) * 256 + k0], ah);
        v16bf bf_ = bfragp(bptr(Wh1, NT, k0 >> 5, nb, lane));
        acc = __builtin_amdgcn_wmma_f32_16x16x32_bf16(false, af, false, bf_, (short)0, acc, false, false);
      }
      float bv = b1[ncol];
#pragma unroll
      for (int v = 0; v < 8; ++v) {
        int mloc = ml + v;
        g[mloc * 1024 + ncol] = acc[v] + bv;
      }
    }
    __syncthreads();
    // ---- h1/c1 update ----
    for (int i = tid; i < 16 * 256; i += 256) {
      int m = i >> 8, j = i & 255;
      float gi = sigmf(g[m * 1024 + j]);
      float gf = sigmf(g[m * 1024 + 256 + j]);
      float gg = tanhf(g[m * 1024 + 512 + j]);
      float go = sigmf(g[m * 1024 + 768 + j]);
      float c = gf * c1[i] + gi * gg;
      c1[i] = c;
      h1[i] = f2bf(go * tanhf(c));
    }
    __syncthreads();
    // ---- output head ----
    if (tid < 32) {
      int m = tid >> 1, k = tid & 1;
      float acc = opb[k];
      for (int j = 0; j < 256; ++j) acc += bf2f(h1[m * 256 + j]) * opW[j * 2 + k];
      int gm = mbase + m;
      if (gm < M) out[((size_t)gm * OUT_LEN + t) * 2 + k] = acc;
    }
    __syncthreads();
  }
}

// ---------------------------------------------------------------------------
// Host orchestration
// ---------------------------------------------------------------------------
extern "C" void kernel_launch(void* const* d_in, const int* in_sizes, int n_in,
                              void* d_out, int out_size, void* d_ws, size_t ws_size,
                              hipStream_t stream) {
  const float* x    = (const float*)d_in[0];
  const int*   ei   = (const int*)  d_in[1];
  const int*   tgt  = (const int*)  d_in[2];
  const float* ipW  = (const float*)d_in[3];
  const float* ipb  = (const float*)d_in[4];
  const float* gWx  = (const float*)d_in[5];
  const float* gWh  = (const float*)d_in[6];
  const float* gbx  = (const float*)d_in[7];
  const float* gbh  = (const float*)d_in[8];
  const float* dynW = (const float*)d_in[9];
  const float* dynB = (const float*)d_in[10];
  const float* g1W  = (const float*)d_in[11];
  const float* g1as = (const float*)d_in[12];
  const float* g1ad = (const float*)d_in[13];
  const float* g1b  = (const float*)d_in[14];
  const float* g2W  = (const float*)d_in[15];
  const float* g2as = (const float*)d_in[16];
  const float* g2ad = (const float*)d_in[17];
  const float* g2b  = (const float*)d_in[18];
  const float* fcW  = (const float*)d_in[19];
  const float* fcb  = (const float*)d_in[20];
  const float* l0Wx = (const float*)d_in[21];
  const float* l0Wh = (const float*)d_in[22];
  const float* l0b  = (const float*)d_in[23];
  const float* l1Wx = (const float*)d_in[24];
  const float* l1Wh = (const float*)d_in[25];
  const float* l1b  = (const float*)d_in[26];
  const float* opW  = (const float*)d_in[27];
  const float* opb  = (const float*)d_in[28];
  float* out = (float*)d_out;

  const int N  = in_sizes[0] / (HIST * 2);   // 30000 nodes
  const int E  = in_sizes[1] / 2;            // 300000 edges
  const int T  = in_sizes[2];                // 3000 targets
  const int ET = E + N;                      // + self-loops
  const int Mp = (T + 15) & ~15;             // padded target rows (3008)
  const int F  = HEADS * H_ENC;              // 384

  // ---- workspace carve ----
  size_t off = 0;
  auto carve = [&](size_t bytes) -> void* {
    void* p = (char*)d_ws + off;
    off += (bytes + 255) & ~(size_t)255;
    return p;
  };
  __bf16* wGx  = (__bf16*)carve(2u * 32 * 384);
  __bf16* wGh  = (__bf16*)carve(2u * 128 * 384);
  __bf16* wDyn = (__bf16*)carve(2u * 128 * 128);
  __bf16* wG1  = (__bf16*)carve(2u * 128 * 384);
  __bf16* wG2  = (__bf16*)carve(2u * 384 * 384);
  __bf16* wFc  = (__bf16*)carve(2u * 384 * 128);
  __bf16* wL0x = (__bf16*)carve(2u * 128 * 1024);
  __bf16* wL0h = (__bf16*)carve(2u * 256 * 1024);
  __bf16* wL1x = (__bf16*)carve(2u * 256 * 1024);
  __bf16* wL1h = (__bf16*)carve(2u * 256 * 1024);
  __bf16* hleak   = (__bf16*)carve(2ull * N * 128);
  __bf16* histbf  = (__bf16*)carve(2ull * N * 128);
  float*  hfeat   = (float*) carve(4ull * N * F);
  float*  ssrc    = (float*) carve(4ull * N * HEADS);
  float*  sdst    = (float*) carve(4ull * N * HEADS);
  int*    maxb    = (int*)   carve(4ull * N * HEADS);
  float*  denom   = (float*) carve(4ull * N * HEADS);
  float*  exbuf   = (float*) carve(4ull * ET * HEADS);
  float*  gat1    = (float*) carve(4ull * N * F);
  __bf16* gat1bf  = (__bf16*)carve(2ull * N * F);
  float*  gat2    = (float*) carve(4ull * N * F);
  __bf16* gatherA = (__bf16*)carve(2ull * Mp * F);
  __bf16* encbf   = (__bf16*)carve(2ull * Mp * 128);
  float*  gx0     = (float*) carve(4ull * Mp * 1024);
  (void)ws_size; (void)n_in; (void)out_size;

  auto pack = [&](const float* s, __bf16* d, int K, int Nn) {
    int n = K * Nn;
    pack_b_kernel<<<(n + 255) / 256, 256, 0, stream>>>(s, d, K, Nn);
  };
  // ---- weights -> bf16, fragment-packed (L2-resident: ~2.4 MB total) ----
  pack(gWx,  wGx,  32, 384);
  pack(gWh,  wGh,  128, 384);
  pack(dynW, wDyn, 128, 128);
  pack(g1W,  wG1,  128, 384);
  pack(g2W,  wG2,  384, 384);
  pack(fcW,  wFc,  384, 128);
  pack(l0Wx, wL0x, 128, 1024);
  pack(l0Wh, wL0h, 256, 1024);
  pack(l1Wx, wL1x, 256, 1024);
  pack(l1Wh, wL1h, 256, 1024);

  // ---- encoder: embed + GRU (fused) ----
  gru_kernel<<<(N + 15) / 16, 256, 0, stream>>>(x, ipW, ipb, wGx, wGh, gbx, gbh, hleak, N);

  // ---- dyn_emb: hist_enc = leaky(leaky(h) @ dyn_W + dyn_b) ----
  gemm_bf16<1><<<dim3((N + 15) / 16, 1), 256, 0, stream>>>(
      hleak, 128, wDyn, dynB, (float*)nullptr, histbf, N, 128, 128);

  // ---- GAT layer helper ----
  auto gat = [&](const float* aS, const float* aD, const float* bias,
                 float* feat, float* outp, __bf16* outbf) {
    int nh = N * HEADS, eh = ET * HEADS;
    gat_scores_kernel<<<(nh + 255) / 256, 256, 0, stream>>>(feat, aS, aD, ssrc, sdst, N);
    fill_ord_kernel<<<(nh + 255) / 256, 256, 0, stream>>>(maxb, -3.0e38f, nh);
    hipMemsetAsync(denom, 0, sizeof(float) * (size_t)nh, stream);
    hipMemsetAsync(outp, 0, sizeof(float) * (size_t)N * F, stream);
    edge_max_kernel<<<(eh + 255) / 256, 256, 0, stream>>>(ei, E, ET, ssrc, sdst, maxb);
    edge_exp_kernel<<<(eh + 255) / 256, 256, 0, stream>>>(ei, E, ET, ssrc, sdst, maxb, exbuf, denom);
    edge_msg_kernel<<<(eh + 255) / 256, 256, 0, stream>>>(ei, E, ET, feat, exbuf, denom, outp);
    int tot = N * F;
    bias_bf_kernel<<<(tot + 255) / 256, 256, 0, stream>>>(outp, bias, outbf, tot, F);
  };

  // ---- GAT 1: hfeat = hist_enc @ g1_W ; attention -> gat1 (+bf16) ----
  gemm_bf16<0><<<dim3((N + 15) / 16, F / 128), 256, 0, stream>>>(
      histbf, 128, wG1, (const float*)nullptr, hfeat, (__bf16*)nullptr, N, F, 128);
  gat(g1as, g1ad, g1b, hfeat, gat1, gat1bf);

  // ---- GAT 2: hfeat = gat1 @ g2_W ; attention -> gat2 ----
  gemm_bf16<0><<<dim3((N + 15) / 16, F / 128), 256, 0, stream>>>(
      gat1bf, F, wG2, (const float*)nullptr, hfeat, (__bf16*)nullptr, N, F, F);
  gat(g2as, g2ad, g2b, hfeat, gat2, (__bf16*)nullptr);

  // ---- gather targets (zero-padded to Mp rows) + fc -> enc (bf16) ----
  gather_pad_kernel<<<(Mp * F + 255) / 256, 256, 0, stream>>>(gat2, tgt, gatherA, T, Mp, F);
  gemm_bf16<1><<<dim3(Mp / 16, 1), 256, 0, stream>>>(
      gatherA, F, wFc, fcb, (float*)nullptr, encbf, Mp, 128, F);

  // ---- precompute gx0 = enc @ l0_Wx + l0_b (constant across decoder steps) ----
  gemm_bf16<0><<<dim3(Mp / 16, 1024 / 128), 256, 0, stream>>>(
      encbf, 128, wL0x, l0b, gx0, (__bf16*)nullptr, Mp, 1024, 128);

  // ---- fused 2-layer LSTM decoder + output head ----
  const size_t dec_lds = 16 * 1024 * 4 + 2 * (16 * 256 * 4) + 2 * (16 * 256 * 2); // 114688 B
  decoder_kernel<<<Mp / 16, 256, dec_lds, stream>>>(
      gx0, wL0h, wL1x, wL1h, l1b, opW, opb, out, T);
}
